// RPN_43834436223770
// MI455X (gfx1250) — compile-verified, matrix operand order checked by reading, not run on hardware
//
#include <hip/hip_runtime.h>
#include <hip/hip_bf16.h>

// ---------------------------------------------------------------------------
// RPN loss for MI455X (gfx1250).  Conv3x3 as implicit GEMM on
// v_wmma_f32_16x16x32_bf16 with hi/lo bf16 split (~f32 accuracy, 3 WMMA / K32).
// Feature map pre-transposed to NHWC hi/lo bf16 planes so every fragment is
// pure b128 loads; M=64 x N=32 tile per wave (24 WMMA / K-block).
// Fragments are built as pure SSA vector values (bit_cast + shufflevector) so
// no alloca exists for PromoteAlloca to push into LDS.
// ---------------------------------------------------------------------------

typedef __attribute__((ext_vector_type(16))) __bf16   v16bf;
typedef __attribute__((ext_vector_type(8)))  __bf16   v8bf;
typedef __attribute__((ext_vector_type(8)))  float    v8f;
typedef __attribute__((ext_vector_type(4)))  unsigned u32x4;

__device__ __forceinline__ v16bf catbf(u32x4 a, u32x4 b) {
    v8bf lo = __builtin_bit_cast(v8bf, a);
    v8bf hi = __builtin_bit_cast(v8bf, b);
    return __builtin_shufflevector(lo, hi, 0, 1, 2, 3, 4, 5, 6, 7,
                                           8, 9, 10, 11, 12, 13, 14, 15);
}

// unconditional fragment: two aligned b128 loads (p points at c0+kBase element)
__device__ __forceinline__ v16bf loadFragU(const unsigned short* p) {
    u32x4 a = *(const u32x4*)(p);
    u32x4 b = *(const u32x4*)(p + 16);
    return catbf(a, b);
}

// guarded fragment: zeros outside the padded halo
__device__ __forceinline__ v16bf loadFragC(const unsigned short* p, bool ok) {
    u32x4 z = (u32x4)(0u);
    u32x4 a = ok ? *(const u32x4*)(p) : z;
    u32x4 b = ok ? *(const u32x4*)(p + 16) : z;
    return catbf(a, b);
}

constexpr int A_TOTAL = 159882;          // 3 * (13^2+25^2+50^2+100^2+200^2)
constexpr int CIN = 256, COUT = 256;

constexpr size_t alignup(size_t x) { return (x + 255) & ~(size_t)255; }

// workspace layout (bytes)
constexpr size_t OFF_XBUF   = 0;                                              // 2*256*200*200 f32
constexpr size_t OFF_LOGITS = OFF_XBUF   + alignup((size_t)2*256*200*200*4);  // 2*A f32
constexpr size_t OFF_DELTAS = OFF_LOGITS + alignup((size_t)2*A_TOTAL*4);      // 2*A*4 f32
constexpr size_t OFF_MVAL   = OFF_DELTAS + alignup((size_t)2*A_TOTAL*16);
constexpr size_t OFF_MATCH  = OFF_MVAL   + alignup((size_t)2*A_TOTAL*4);
constexpr size_t OFF_LABEL  = OFF_MATCH  + alignup((size_t)2*A_TOTAL*4);
constexpr size_t OFF_BCE    = OFF_LABEL  + alignup((size_t)2*A_TOTAL*4);
constexpr size_t OFF_L1     = OFF_BCE    + alignup((size_t)2*A_TOTAL*4);
constexpr size_t OFF_BESTG  = OFF_L1     + alignup((size_t)2*A_TOTAL*4);      // 40 int
constexpr size_t OFF_WHI    = OFF_BESTG  + alignup((size_t)40*4);             // 9*256*256 bf16
constexpr size_t OFF_WLO    = OFF_WHI    + alignup((size_t)9*256*256*2);
constexpr size_t OFF_PART   = OFF_WLO    + alignup((size_t)9*256*256*2);      // 4 f32
constexpr size_t OFF_FMHI   = OFF_PART   + alignup((size_t)4*4);              // 2*200*200*256 bf16
constexpr size_t OFF_FMLO   = OFF_FMHI   + alignup((size_t)2*200*200*256*2);

// ---------------------------------------------------------------------------
__global__ void initK(int* bestG) {
    int i = threadIdx.x;
    if (i < 40) bestG[i] = 0;   // 0 bits == 0.0f, valid since IoU >= 0
}

// w_in [ko][c][3][3] f32  ->  wHi/wLo [rs][ko][c] bf16 (hi/lo split)
__global__ void splitW(const float* __restrict__ w_in,
                       unsigned short* __restrict__ wHi, unsigned short* __restrict__ wLo) {
    int i = blockIdx.x * 256 + threadIdx.x;
    if (i >= 9 * 256 * 256) return;
    int rs = i >> 16;
    int ko = (i >> 8) & 255;
    int c  = i & 255;
    float v = w_in[((size_t)ko * 256 + c) * 9 + rs];
    __bf16 h = (__bf16)v;
    __bf16 l = (__bf16)(v - (float)h);
    wHi[i] = __builtin_bit_cast(unsigned short, h);
    wLo[i] = __builtin_bit_cast(unsigned short, l);
}

// fm [2][256][HW] f32 -> hiT/loT [2][HW][256] bf16-bits (NHWC), LDS transpose.
// block 256 = 32(x) x 8(y); tile = 32 channels x 32 pixels.
__global__ void packFMT(const float* __restrict__ fm,
                        unsigned short* __restrict__ hiT, unsigned short* __restrict__ loT,
                        int HW) {
    const int b  = blockIdx.z;
    const int c0 = blockIdx.y * 32;
    const int p0 = blockIdx.x * 32;
    __shared__ unsigned short sH[32][33];
    __shared__ unsigned short sL[32][33];
    const int tx = threadIdx.x & 31, ty = threadIdx.x >> 5;

#pragma unroll
    for (int it = 0; it < 4; ++it) {
        int cl = it * 8 + ty;                 // local channel
        int p  = p0 + tx;                     // pixel (coalesced reads)
        float v = (p < HW) ? fm[((size_t)b * CIN + c0 + cl) * HW + p] : 0.0f;
        __bf16 h = (__bf16)v;
        __bf16 l = (__bf16)(v - (float)h);
        sH[cl][tx] = __builtin_bit_cast(unsigned short, h);
        sL[cl][tx] = __builtin_bit_cast(unsigned short, l);
    }
    __syncthreads();
#pragma unroll
    for (int it = 0; it < 4; ++it) {
        int pl = it * 8 + ty;                 // local pixel
        int p  = p0 + pl;
        if (p < HW) {                         // coalesced writes along c
            size_t o = ((size_t)b * HW + p) * CIN + c0 + tx;
            hiT[o] = sH[tx][pl];
            loT[o] = sL[tx][pl];
        }
    }
}

// ---------------------------------------------------------------------------
// Conv3x3 (pad=1) + bias + ReLU as implicit GEMM.
// One wave per M=64 (4 WMMA tiles) x N=32 (2 subtiles) output tile.
// K loop: rs outer (9), c inner in blocks of 32; 24 WMMAs per K-block.
__global__ __launch_bounds__(32)
void conv3x3_wmma(const unsigned short* __restrict__ fmHiT,
                  const unsigned short* __restrict__ fmLoT,
                  const unsigned short* __restrict__ wHi,
                  const unsigned short* __restrict__ wLo,
                  const float* __restrict__ bias,
                  float* __restrict__ xbuf, int H, int W) {
    const int lane = threadIdx.x;
    const int n0 = blockIdx.x * 32;          // spatial tile (2 x 16)
    const int m0 = blockIdx.y * 64;          // output channel tile (4 x 16)
    const int HW = H * W, BHW = 2 * HW;

    // CDNA5 16-bit A/B fragment K mapping: element e -> K = (e<8?e:e+8)+kBase
    const int kBase = (lane < 16) ? 0 : 8;
    const int mLane = lane & 15;

    // subtile 0
    const int nIdx0 = n0 + (lane & 15);
    const bool pvalid0 = nIdx0 < BHW;
    int pb0 = 0, py0 = 0, px0 = 0;
    if (pvalid0) { pb0 = nIdx0 / HW; int r = nIdx0 % HW; py0 = r / W; px0 = r % W; }
    // subtile 1
    const int nIdx1 = nIdx0 + 16;
    const bool pvalid1 = nIdx1 < BHW;
    int pb1 = 0, py1 = 0, px1 = 0;
    if (pvalid1) { pb1 = nIdx1 / HW; int r = nIdx1 % HW; py1 = r / W; px1 = r % W; }

    v8f acc[4][2];
#pragma unroll
    for (int t = 0; t < 4; ++t) {
        acc[t][0] = (v8f){};
        acc[t][1] = (v8f){};
    }

    for (int rs = 0; rs < 9; ++rs) {
        const int dy = rs / 3 - 1, dx = rs % 3 - 1;
        const unsigned short* ahBase = wHi + ((size_t)rs * COUT + m0 + mLane) * CIN + kBase;
        const unsigned short* alBase = wLo + ((size_t)rs * COUT + m0 + mLane) * CIN + kBase;

        const int yy0 = py0 + dy, xx0 = px0 + dx;
        const bool ok0 = pvalid0 && (yy0 >= 0) && (yy0 < H) && (xx0 >= 0) && (xx0 < W);
        const unsigned short* hp0 =
            fmHiT + ((size_t)pb0 * HW + (size_t)yy0 * W + xx0) * CIN + kBase;
        const unsigned short* lp0 =
            fmLoT + ((size_t)pb0 * HW + (size_t)yy0 * W + xx0) * CIN + kBase;

        const int yy1 = py1 + dy, xx1 = px1 + dx;
        const bool ok1 = pvalid1 && (yy1 >= 0) && (yy1 < H) && (xx1 >= 0) && (xx1 < W);
        const unsigned short* hp1 =
            fmHiT + ((size_t)pb1 * HW + (size_t)yy1 * W + xx1) * CIN + kBase;
        const unsigned short* lp1 =
            fmLoT + ((size_t)pb1 * HW + (size_t)yy1 * W + xx1) * CIN + kBase;

        for (int c0 = 0; c0 < CIN; c0 += 32) {
            // ---- B fragments: pure b128 loads, SSA values only ----
            v16bf bH0 = loadFragC(hp0 + c0, ok0);
            v16bf bL0 = loadFragC(lp0 + c0, ok0);
            v16bf bH1 = loadFragC(hp1 + c0, ok1);
            v16bf bL1 = loadFragC(lp1 + c0, ok1);
            // ---- 4 M-tiles: b128 A loads + 6 WMMAs per tile ----
#pragma unroll
            for (int t = 0; t < 4; ++t) {
                v16bf aH = loadFragU(ahBase + (size_t)t * 16 * CIN + c0);
                v16bf aL = loadFragU(alBase + (size_t)t * 16 * CIN + c0);
                acc[t][0] = __builtin_amdgcn_wmma_f32_16x16x32_bf16(
                    false, aH, false, bH0, (short)0, acc[t][0], false, false);
                acc[t][0] = __builtin_amdgcn_wmma_f32_16x16x32_bf16(
                    false, aH, false, bL0, (short)0, acc[t][0], false, false);
                acc[t][0] = __builtin_amdgcn_wmma_f32_16x16x32_bf16(
                    false, aL, false, bH0, (short)0, acc[t][0], false, false);
                acc[t][1] = __builtin_amdgcn_wmma_f32_16x16x32_bf16(
                    false, aH, false, bH1, (short)0, acc[t][1], false, false);
                acc[t][1] = __builtin_amdgcn_wmma_f32_16x16x32_bf16(
                    false, aH, false, bL1, (short)0, acc[t][1], false, false);
                acc[t][1] = __builtin_amdgcn_wmma_f32_16x16x32_bf16(
                    false, aL, false, bH1, (short)0, acc[t][1], false, false);
            }
        }
    }

    const int mAdd = (lane < 16) ? 0 : 8;   // C/D layout: VGPR v, lanes>=16 -> M=v+8
    if (pvalid0) {
        const size_t pixOut = (size_t)py0 * W + px0;
#pragma unroll
        for (int t = 0; t < 4; ++t) {
#pragma unroll
            for (int v = 0; v < 8; ++v) {
                const int ko = m0 + t * 16 + v + mAdd;
                float val = acc[t][0][v] + bias[ko];
                val = val > 0.0f ? val : 0.0f;
                xbuf[((size_t)pb0 * COUT + ko) * HW + pixOut] = val;
            }
        }
    }
    if (pvalid1) {
        const size_t pixOut = (size_t)py1 * W + px1;
#pragma unroll
        for (int t = 0; t < 4; ++t) {
#pragma unroll
            for (int v = 0; v < 8; ++v) {
                const int ko = m0 + t * 16 + v + mAdd;
                float val = acc[t][1][v] + bias[ko];
                val = val > 0.0f ? val : 0.0f;
                xbuf[((size_t)pb1 * COUT + ko) * HW + pixOut] = val;
            }
        }
    }
}

// ---------------------------------------------------------------------------
// 1x1 heads: 3 logits + 12 deltas per pixel, written in the reference's
// transposed + level-concatenated anchor order.
__global__ void heads1x1(const float* __restrict__ xbuf,
                         const float* __restrict__ w_obj, const float* __restrict__ b_obj,
                         const float* __restrict__ w_del, const float* __restrict__ b_del,
                         float* __restrict__ logits, float* __restrict__ deltas,
                         int H, int W, int levOff) {
    const int HW = H * W, BHW = 2 * HW;
    int p = blockIdx.x * 256 + threadIdx.x;
    if (p >= BHW) return;
    int b = p / HW, r = p % HW;

    float acc[15];
#pragma unroll
    for (int j = 0; j < 15; ++j) acc[j] = 0.0f;

    const float* xp = xbuf + (size_t)b * COUT * HW + r;
    for (int c = 0; c < CIN; ++c) {
        float v = xp[(size_t)c * HW];
#pragma unroll
        for (int j = 0; j < 3; ++j)  acc[j]     += v * w_obj[j * 256 + c];
#pragma unroll
        for (int j = 0; j < 12; ++j) acc[3 + j] += v * w_del[j * 256 + c];
    }

    const int base = b * A_TOTAL + levOff + r * 3;   // anchor index of (pixel, a=0)
#pragma unroll
    for (int a = 0; a < 3; ++a) {
        logits[base + a] = acc[a] + b_obj[a];
#pragma unroll
        for (int d = 0; d < 4; ++d)
            deltas[(size_t)(base + a) * 4 + d] = acc[3 + a * 4 + d] + b_del[a * 4 + d];
    }
}

// ---------------------------------------------------------------------------
__device__ __forceinline__ float iou_fn(float ax0, float ay0, float ax1, float ay1,
                                        float aArea, const float* g) {
    float gx0 = g[0], gy0 = g[1], gx1 = g[2], gy1 = g[3];
    float gArea = (gx1 - gx0) * (gy1 - gy0);
    float ix0 = fmaxf(ax0, gx0), iy0 = fmaxf(ay0, gy0);
    float ix1 = fminf(ax1, gx1), iy1 = fminf(ay1, gy1);
    float iw = fmaxf(ix1 - ix0, 0.0f), ih = fmaxf(iy1 - iy0, 0.0f);
    float inter = iw * ih;
    return inter / (aArea + gArea - inter);
}

// pass 1: per-anchor max/argmax IoU; per-gt global best via int-bits atomicMax
__global__ void matchK(const float* __restrict__ anchors, const float* __restrict__ gts,
                       float* __restrict__ mval, int* __restrict__ matched,
                       int* __restrict__ bestG) {
    const int b = blockIdx.y;
    const int a = blockIdx.x * 256 + threadIdx.x;
    __shared__ int sBest[20];
    if (threadIdx.x < 20) sBest[threadIdx.x] = 0;
    __syncthreads();

    if (a < A_TOTAL) {
        float ax0 = anchors[a * 4 + 0], ay0 = anchors[a * 4 + 1];
        float ax1 = anchors[a * 4 + 2], ay1 = anchors[a * 4 + 3];
        float aArea = (ax1 - ax0) * (ay1 - ay0);
        float best = -1.0f; int bi = 0;
        for (int g = 0; g < 20; ++g) {
            float iou = iou_fn(ax0, ay0, ax1, ay1, aArea, gts + ((size_t)b * 20 + g) * 4);
            if (iou > best) { best = iou; bi = g; }          // ties keep first (== argmax)
            atomicMax(&sBest[g], __float_as_int(iou));        // IoU>=0: bit order == fp order
        }
        mval[(size_t)b * A_TOTAL + a] = best;
        matched[(size_t)b * A_TOTAL + a] = bi;
    }
    __syncthreads();
    if (threadIdx.x < 20) atomicMax(&bestG[b * 20 + threadIdx.x], sBest[threadIdx.x]);
}

// pass 2: labels (incl. low-quality matches), per-anchor BCE and L1
__global__ void labelK(const float* __restrict__ anchors, const float* __restrict__ gts,
                       const float* __restrict__ mval, const int* __restrict__ matched,
                       const int* __restrict__ bestG,
                       const float* __restrict__ logits, const float* __restrict__ deltas,
                       float* __restrict__ labelA, float* __restrict__ bceA,
                       float* __restrict__ l1A) {
    const int b = blockIdx.y;
    const int a = blockIdx.x * 256 + threadIdx.x;
    if (a >= A_TOTAL) return;
    const size_t idx = (size_t)b * A_TOTAL + a;

    float ax0 = anchors[a * 4 + 0], ay0 = anchors[a * 4 + 1];
    float ax1 = anchors[a * 4 + 2], ay1 = anchors[a * 4 + 3];
    float aArea = (ax1 - ax0) * (ay1 - ay0);

    float mv = mval[idx];
    float label = (mv >= 0.7f) ? 1.0f : ((mv < 0.3f) ? 0.0f : -1.0f);
    bool lq = false;
    for (int g = 0; g < 20; ++g) {
        float iou = iou_fn(ax0, ay0, ax1, ay1, aArea, gts + ((size_t)b * 20 + g) * 4);
        float bg = __int_as_float(bestG[b * 20 + g]);
        if (iou == bg && bg > 0.0f) lq = true;
    }
    if (lq) label = 1.0f;
    labelA[idx] = label;

    // BCE with logits (target = clip(label,0,1))
    float z = logits[idx];
    float t = (label > 0.0f) ? 1.0f : 0.0f;
    bceA[idx] = fmaxf(z, 0.0f) - z * t + log1pf(expf(-fabsf(z)));

    // L1 on deltas vs get_deltas(anchor, matched gt)
    const float* gb = gts + ((size_t)b * 20 + matched[idx]) * 4;
    float sw = ax1 - ax0, sh = ay1 - ay0;
    float scx = ax0 + 0.5f * sw, scy = ay0 + 0.5f * sh;
    float tw = gb[2] - gb[0], th = gb[3] - gb[1];
    float tcx = gb[0] + 0.5f * tw, tcy = gb[1] + 0.5f * th;
    float gd0 = (tcx - scx) / sw, gd1 = (tcy - scy) / sh;
    float gd2 = logf(tw / sw),    gd3 = logf(th / sh);
    const float* dl = deltas + idx * 4;
    l1A[idx] = fabsf(dl[0] - gd0) + fabsf(dl[1] - gd1) +
               fabsf(dl[2] - gd2) + fabsf(dl[3] - gd3);
}

// ---------------------------------------------------------------------------
// sequential subsample (cumsum semantics) + loss sums; one wave32 per image.
__global__ __launch_bounds__(32)
void sampleLossK(const float* __restrict__ labelA, const float* __restrict__ bceA,
                 const float* __restrict__ l1A, float* __restrict__ partial) {
    const int b = blockIdx.x;
    const int lane = threadIdx.x;
    const float* L  = labelA + (size_t)b * A_TOTAL;
    const float* BC = bceA   + (size_t)b * A_TOTAL;
    const float* L1 = l1A    + (size_t)b * A_TOTAL;
    const unsigned lmaskLt = (lane == 0) ? 0u : ((1u << lane) - 1u);
    const int iters = (A_TOTAL + 31) / 32;

    float objSum = 0.0f, locSum = 0.0f;

    // phase 1: positives — sel_pos = pos & cumsum(pos) <= 128
    int posCount = 0;
    for (int it = 0; it < iters; ++it) {
        int a = it * 32 + lane;
        bool pos = (a < A_TOTAL) && (L[a] == 1.0f);
        unsigned m = (unsigned)__ballot(pos);
        int rank = __popc(m & lmaskLt);
        if (pos && (posCount + rank + 1 <= 128)) { objSum += BC[a]; locSum += L1[a]; }
        posCount += __popc(m);                     // wave-uniform
        if (posCount >= 128) break;                // n_pos saturates at 128
    }
    int nPos = posCount > 128 ? 128 : posCount;
    int negLimit = 256 - nPos;

    // phase 2: negatives — sel_neg = neg & cumsum(neg) <= 256 - n_pos
    int negCount = 0;
    for (int it = 0; it < iters; ++it) {
        int a = it * 32 + lane;
        bool neg = (a < A_TOTAL) && (L[a] == 0.0f);
        unsigned m = (unsigned)__ballot(neg);
        int rank = __popc(m & lmaskLt);
        if (neg && (negCount + rank + 1 <= negLimit)) objSum += BC[a];
        negCount += __popc(m);
        if (negCount >= negLimit) break;
    }

    __shared__ float sObj[32], sLoc[32];
    sObj[lane] = objSum; sLoc[lane] = locSum;
    __syncthreads();
    if (lane == 0) {
        float o = 0.0f, l = 0.0f;
        for (int i = 0; i < 32; ++i) { o += sObj[i]; l += sLoc[i]; }
        partial[b * 2 + 0] = o;
        partial[b * 2 + 1] = l;
    }
}

__global__ void finalK(const float* __restrict__ partial, float* __restrict__ out) {
    out[0] = 0.5f * (partial[0] + partial[2]);
    out[1] = 0.5f * (partial[1] + partial[3]);
}

// ---------------------------------------------------------------------------
extern "C" void kernel_launch(void* const* d_in, const int* in_sizes, int n_in,
                              void* d_out, int out_size, void* d_ws, size_t ws_size,
                              hipStream_t stream) {
    const float* feats[5] = {(const float*)d_in[0], (const float*)d_in[1],
                             (const float*)d_in[2], (const float*)d_in[3],
                             (const float*)d_in[4]};
    const float* anchors = (const float*)d_in[5];
    const float* gts     = (const float*)d_in[6];
    const float* w_in    = (const float*)d_in[7];
    const float* b_in    = (const float*)d_in[8];
    const float* w_obj   = (const float*)d_in[9];
    const float* b_obj   = (const float*)d_in[10];
    const float* w_del   = (const float*)d_in[11];
    const float* b_del   = (const float*)d_in[12];

    char* ws = (char*)d_ws;
    float*          xbuf    = (float*)(ws + OFF_XBUF);
    float*          logits  = (float*)(ws + OFF_LOGITS);
    float*          deltas  = (float*)(ws + OFF_DELTAS);
    float*          mval    = (float*)(ws + OFF_MVAL);
    int*            matched = (int*)  (ws + OFF_MATCH);
    float*          labelA  = (float*)(ws + OFF_LABEL);
    float*          bceA    = (float*)(ws + OFF_BCE);
    float*          l1A     = (float*)(ws + OFF_L1);
    int*            bestG   = (int*)  (ws + OFF_BESTG);
    unsigned short* wHi     = (unsigned short*)(ws + OFF_WHI);
    unsigned short* wLo     = (unsigned short*)(ws + OFF_WLO);
    float*          partial = (float*)(ws + OFF_PART);
    unsigned short* fmHiT   = (unsigned short*)(ws + OFF_FMHI);
    unsigned short* fmLoT   = (unsigned short*)(ws + OFF_FMLO);

    initK<<<1, 64, 0, stream>>>(bestG);
    splitW<<<(9 * 256 * 256 + 255) / 256, 256, 0, stream>>>(w_in, wHi, wLo);

    static const int levH[5]   = {13, 25, 50, 100, 200};
    static const int levOff[5] = {0, 507, 2382, 9882, 39882};
    for (int l = 0; l < 5; ++l) {
        const int H = levH[l], W = levH[l], HW = H * W, BHW = 2 * HW;
        packFMT<<<dim3((HW + 31) / 32, CIN / 32, 2), 256, 0, stream>>>(
            feats[l], fmHiT, fmLoT, HW);
        const int ntiles = (BHW + 31) / 32;
        conv3x3_wmma<<<dim3(ntiles, COUT / 64), 32, 0, stream>>>(
            fmHiT, fmLoT, wHi, wLo, b_in, xbuf, H, W);
        heads1x1<<<(BHW + 255) / 256, 256, 0, stream>>>(
            xbuf, w_obj, b_obj, w_del, b_del, logits, deltas, H, W, levOff[l]);
    }

    const dim3 ag((A_TOTAL + 255) / 256, 2);
    matchK<<<ag, 256, 0, stream>>>(anchors, gts, mval, matched, bestG);
    labelK<<<ag, 256, 0, stream>>>(anchors, gts, mval, matched, bestG,
                                   logits, deltas, labelA, bceA, l1A);
    sampleLossK<<<2, 32, 0, stream>>>(labelA, bceA, l1A, partial);
    finalK<<<1, 1, 0, stream>>>(partial, (float*)d_out);
}
